// RobertaGCN_36618891166414
// MI455X (gfx1250) — matrix operand order repaired and checked by reading, not compiled, not run on hardware
//
#include <hip/hip_runtime.h>

#define B_  32
#define S_  1024
#define D_  768
#define H1_ 256
#define H2_ 128

typedef __bf16 bf16;
typedef unsigned int u32;
typedef __attribute__((ext_vector_type(16))) __bf16 v16bf;
typedef __attribute__((ext_vector_type(8)))  __bf16 v8bf;
typedef __attribute__((ext_vector_type(8)))  float  v8f;
typedef __attribute__((ext_vector_type(4)))  u32    v4u;
typedef __attribute__((ext_vector_type(8)))  u32    v8u;

static __device__ __forceinline__ v8f v8f_zero() {
  v8f z = {0.f, 0.f, 0.f, 0.f, 0.f, 0.f, 0.f, 0.f};
  return z;
}

// D = A(16x32 bf16) * B(32x16 bf16) + C(16x16 f32)
static __device__ __forceinline__ v8f wmma_bf16(v16bf a, v16bf b, v8f c) {
  return __builtin_amdgcn_wmma_f32_16x16x32_bf16(false, a, false, b, (short)0, c,
                                                 false, false);
}

// WMMA A-operand loader from row-major global memory: lane's 16 elems are two
// 16B runs (K = kb..kb+7 and kb+16..kb+23). p points at run 0.
static __device__ __forceinline__ v16bf load_a16(const bf16* p) {
  union { v16bf v; v8bf h[2]; } u;
  u.h[0] = *(const v8bf*)(p);
  u.h[1] = *(const v8bf*)(p + 16);
  return u.v;
}

// ---------------------------------------------------------------------------
// TDM: DMA a [16 x rowlen] bf16 stripe (row stride = rowlen) from global into
// LDS with dword padding (pad_amount=4 dwords every 2^pad interval) to break
// LDS bank conflicts on the transposed A-operand reads.
//   padIntervalCode: 6 -> every 128 dwords, 7 -> every 256 dwords
// Issued by one wave; caller must __syncthreads() afterwards.
// ---------------------------------------------------------------------------
static __device__ __forceinline__ void tdm_load_stripe16(const bf16* gsrc,
                                                         bf16* lds_dst,
                                                         u32 rowlen,
                                                         u32 padIntervalCode) {
  const unsigned long long ga = (unsigned long long)(size_t)gsrc;
  const u32 lb = (u32)(size_t)(void*)lds_dst;   // low 32 bits = LDS byte addr
  v4u g0;
  g0[0] = 1u;                                    // count=1, user descriptor
  g0[1] = lb;                                    // lds_addr (bytes)
  g0[2] = (u32)ga;                               // global_addr[31:0]
  g0[3] = (u32)(ga >> 32) | 0x80000000u;         // addr[56:32] | type=2<<30
  v8u g1;
  g1[0] = (1u << 16)                             // data_size = 2 bytes
        | (1u << 20)                             // pad_enable
        | (padIntervalCode << 22)                // pad_interval
        | (3u << 25);                            // pad_amount = 4 dwords
  g1[1] = (rowlen & 0xFFFFu) << 16;              // tensor_dim0[15:0]
  g1[2] = (rowlen >> 16) | (16u << 16);          // dim0[31:16] | tensor_dim1
  g1[3] = (rowlen << 16);                        // dim1[31:16]=0 | tile_dim0
  g1[4] = 16u;                                   // tile_dim1=16, tile_dim2=0
  g1[5] = rowlen;                                // tensor_dim0_stride[31:0]
  g1[6] = 0u;                                    // stride hi | dim1_stride lo
  g1[7] = 0u;
  asm volatile("tensor_load_to_lds %0, %1" :: "s"(g0), "s"(g1) : "memory");
  __builtin_amdgcn_s_wait_tensorcnt(0);
}

// Padded-LDS addressing for the stripes above (e = even bf16 element index).
// D=768 rows: pad 4 dw / 128 dw -> row stride 396 dw (792 bf16)
static __device__ __forceinline__ const bf16* lds_row768(const bf16* base,
                                                         int r, int e) {
  int dw = e >> 1;
  int dwp = dw + ((dw >> 7) << 2);
  return base + r * 792 + (dwp << 1);
}
// S=1024 rows: pad 4 dw / 256 dw -> row stride 520 dw (1040 bf16)
static __device__ __forceinline__ const bf16* lds_row1024(const bf16* base,
                                                          int r, int e) {
  int dw = e >> 1;
  int dwp = dw + ((dw >> 8) << 2);
  return base + r * 1040 + (dwp << 1);
}

// ---------------- prep kernels ----------------

__global__ void k_cast(const float* __restrict__ x, bf16* __restrict__ y, long n) {
  long i = (long)blockIdx.x * blockDim.x + threadIdx.x;
  long st = (long)gridDim.x * blockDim.x;
  for (; i < n; i += st) y[i] = (bf16)x[i];
}

__global__ void k_w1t(const float* __restrict__ w, bf16* __restrict__ wt) {
  int i = blockIdx.x * blockDim.x + threadIdx.x;     // w: [D,H1] -> wt: [H1,D]
  if (i < D_ * H1_) {
    int d = i / H1_, f = i % H1_;
    wt[(size_t)f * D_ + d] = (bf16)w[i];
  }
}

__global__ void k_w2t(const float* __restrict__ w, bf16* __restrict__ wt) {
  int i = blockIdx.x * blockDim.x + threadIdx.x;     // w: [H1,H2] -> wt: [H2,H1]
  if (i < H1_ * H2_) {
    int d = i / H2_, f = i % H2_;
    wt[(size_t)f * H1_ + d] = (bf16)w[i];
  }
}

__global__ void k_zero(float* __restrict__ p, int n) {
  int i = blockIdx.x * blockDim.x + threadIdx.x;
  if (i < n) p[i] = 0.f;
}

// ---------------- attention: AT[b,t,s] = softmax_t(X Xt)[s,t] ----------------
// Workgroup: rows s0..s0+15, all 1024 cols; wave w owns cols [w*128, w*128+128).
// A-operand stripe (16 rows of X) is TDM-staged in LDS, shared by all 8 waves.

__global__ __launch_bounds__(256) void k_attn(const bf16* __restrict__ Xh,
                                              bf16* __restrict__ AT) {
  const int b    = blockIdx.y;
  const int s0   = blockIdx.x * 16;
  const int wv   = threadIdx.x >> 5;
  const int ln   = threadIdx.x & 31;
  const int l16  = ln & 15;
  const int half = ln >> 4;
  const int kbA  = half ? 16 : 0;
  const int kbB  = half ? 16 : 0;
  const int tbase = wv * 128;

  __shared__ __align__(16) bf16 x_lds[16 * 792];   // padded 16 x D stripe
  __shared__ float red_max[8][16];
  __shared__ float red_sum[8][16];

  if (wv == 0)
    tdm_load_stripe16(Xh + ((size_t)b * S_ + s0) * D_, x_lds, D_, 6u);

  const bf16* bptr[8];
  for (int T = 0; T < 8; ++T)
    bptr[T] = Xh + ((size_t)b * S_ + tbase + T * 16 + l16) * D_ + kbB;

  v8f acc[8];
  for (int T = 0; T < 8; ++T) acc[T] = v8f_zero();

  __syncthreads();   // TDM data visible

  for (int d0 = 0; d0 < D_; d0 += 32) {
    union { v16bf v; v8bf h[2]; } ua;
    ua.h[0] = *(const v8bf*)lds_row768(x_lds, l16, d0 + kbA);
    ua.h[1] = *(const v8bf*)lds_row768(x_lds, l16, d0 + kbA + 8);
    for (int T = 0; T < 8; ++T) {
      v16bf bb = *(const v16bf*)(bptr[T] + d0);
      acc[T] = wmma_bf16(ua.v, bb, acc[T]);
    }
  }

  // per-row max (row = r + 8*half lives across the 16 lanes of one half)
  float gmax[8];
  {
    float rmax[8];
    for (int r = 0; r < 8; ++r) {
      float m = acc[0][r];
      for (int T = 1; T < 8; ++T) m = fmaxf(m, acc[T][r]);
      m = fmaxf(m, __shfl_xor(m, 1));
      m = fmaxf(m, __shfl_xor(m, 2));
      m = fmaxf(m, __shfl_xor(m, 4));
      m = fmaxf(m, __shfl_xor(m, 8));
      rmax[r] = m;
    }
    if (l16 == 0)
      for (int r = 0; r < 8; ++r) red_max[wv][r + 8 * half] = rmax[r];
    __syncthreads();
    for (int r = 0; r < 8; ++r) {
      float m = -3.0e38f;
      for (int w = 0; w < 8; ++w) m = fmaxf(m, red_max[w][r + 8 * half]);
      gmax[r] = m;
    }
  }

  // exp + per-row sum
  float rsum[8];
  for (int r = 0; r < 8; ++r) rsum[r] = 0.f;
  for (int T = 0; T < 8; ++T)
    for (int r = 0; r < 8; ++r) {
      float p = __expf(acc[T][r] - gmax[r]);
      acc[T][r] = p;
      rsum[r] += p;
    }
  for (int r = 0; r < 8; ++r) {
    float s = rsum[r];
    s += __shfl_xor(s, 1);
    s += __shfl_xor(s, 2);
    s += __shfl_xor(s, 4);
    s += __shfl_xor(s, 8);
    rsum[r] = s;
  }
  if (l16 == 0)
    for (int r = 0; r < 8; ++r) red_sum[wv][r + 8 * half] = rsum[r];
  __syncthreads();
  float ginv[8];
  for (int r = 0; r < 8; ++r) {
    float s = 0.f;
    for (int w = 0; w < 8; ++w) s += red_sum[w][r + 8 * half];
    ginv[r] = 1.0f / s;
  }

  // normalize + store transposed: AT[b, t, s] (8 consecutive s per lane = 16B)
  for (int T = 0; T < 8; ++T) {
    v8bf o;
    for (int r = 0; r < 8; ++r) o[r] = (bf16)(acc[T][r] * ginv[r]);
    const size_t t = (size_t)(tbase + T * 16 + l16);
    *(v8bf*)(AT + ((size_t)b * S_ + t) * S_ + s0 + 8 * half) = o;
  }
}

// ---------------- dis[b,t] = rsqrt(sum_s A[s,t]) = rsqrt(rowsum AT) ----------

__global__ __launch_bounds__(256) void k_dis(const bf16* __restrict__ AT,
                                             float* __restrict__ dis) {
  const int b  = blockIdx.y;
  const int t  = blockIdx.x * 8 + (threadIdx.x >> 5);
  const int ln = threadIdx.x & 31;
  const bf16* p = AT + ((size_t)b * S_ + t) * S_;
  float s = 0.f;
  for (int i = ln; i < S_; i += 32) s += (float)p[i];
  s += __shfl_xor(s, 16);
  s += __shfl_xor(s, 8);
  s += __shfl_xor(s, 4);
  s += __shfl_xor(s, 2);
  s += __shfl_xor(s, 1);
  if (ln == 0) dis[b * S_ + t] = rsqrtf(s);
}

// ---------------- G1: P1T[b,f,s] = (dis[s] * (X W1)[s,f]) transposed --------

__global__ __launch_bounds__(256) void k_g1(const bf16* __restrict__ Xh,
                                            const bf16* __restrict__ W1T,
                                            const float* __restrict__ dis,
                                            bf16* __restrict__ P1T) {
  const int b    = blockIdx.y;
  const int s0   = blockIdx.x * 16;
  const int wv   = threadIdx.x >> 5;
  const int ln   = threadIdx.x & 31;
  const int l16  = ln & 15;
  const int half = ln >> 4;
  const int kbA  = half ? 16 : 0;
  const int kbB  = half ? 16 : 0;

  __shared__ __align__(16) bf16 x_lds[16 * 792];
  __shared__ float dls[16];
  if (threadIdx.x < 16) dls[threadIdx.x] = dis[b * S_ + s0 + threadIdx.x];
  if (wv == 0)
    tdm_load_stripe16(Xh + ((size_t)b * S_ + s0) * D_, x_lds, D_, 6u);

  const int f1a = wv * 32 + l16;
  const int f1b = f1a + 16;
  const bf16* bp0 = W1T + (size_t)f1a * D_ + kbB;
  const bf16* bp1 = W1T + (size_t)f1b * D_ + kbB;

  v8f a0 = v8f_zero(), a1 = v8f_zero();
  __syncthreads();
  for (int d0 = 0; d0 < D_; d0 += 32) {
    union { v16bf v; v8bf h[2]; } ua;
    ua.h[0] = *(const v8bf*)lds_row768(x_lds, l16, d0 + kbA);
    ua.h[1] = *(const v8bf*)lds_row768(x_lds, l16, d0 + kbA + 8);
    a0 = wmma_bf16(ua.v, *(const v16bf*)(bp0 + d0), a0);
    a1 = wmma_bf16(ua.v, *(const v16bf*)(bp1 + d0), a1);
  }
  const int sl = 8 * half;
  v8bf o0, o1;
  for (int r = 0; r < 8; ++r) {
    float dv = dls[sl + r];
    o0[r] = (bf16)(dv * a0[r]);
    o1[r] = (bf16)(dv * a1[r]);
  }
  *(v8bf*)(P1T + ((size_t)b * H1_ + f1a) * S_ + s0 + sl) = o0;
  *(v8bf*)(P1T + ((size_t)b * H1_ + f1b) * S_ + s0 + sl) = o1;
}

// ---------------- G2: M1 = AT @ P1 ; h = relu(dis*M1 + b1) (LDS) ;
//                  P2T = (dis * (h W2)) transposed ----------------------------

__global__ __launch_bounds__(256) void k_g2(const bf16* __restrict__ AT,
                                            const bf16* __restrict__ P1T,
                                            const bf16* __restrict__ W2T,
                                            const float* __restrict__ dis,
                                            const float* __restrict__ b1,
                                            bf16* __restrict__ P2T) {
  const int b    = blockIdx.y;
  const int t0   = blockIdx.x * 16;
  const int wv   = threadIdx.x >> 5;
  const int ln   = threadIdx.x & 31;
  const int l16  = ln & 15;
  const int half = ln >> 4;
  const int kbA  = half ? 16 : 0;
  const int kbB  = half ? 16 : 0;

  __shared__ __align__(16) bf16 at_lds[16 * 1040];  // padded 16 x S stripe
  __shared__ __align__(32) bf16 h_lds[16 * 272];    // 16 rows x H1, padded
  __shared__ float dls[16];
  if (threadIdx.x < 16) dls[threadIdx.x] = dis[b * S_ + t0 + threadIdx.x];
  if (wv == 0)
    tdm_load_stripe16(AT + ((size_t)b * S_ + t0) * S_, at_lds, S_, 7u);

  // Phase A: M1 tiles for f1 columns {wv*32, wv*32+16}, K = s over 1024
  const int f1a = wv * 32 + l16;
  const int f1b = f1a + 16;
  const bf16* bp0 = P1T + ((size_t)b * H1_ + f1a) * S_ + kbB;
  const bf16* bp1 = P1T + ((size_t)b * H1_ + f1b) * S_ + kbB;

  v8f a0 = v8f_zero(), a1 = v8f_zero();
  __syncthreads();
  for (int s = 0; s < S_; s += 32) {
    __builtin_prefetch(bp0 + s + 128, 0, 3);
    union { v16bf v; v8bf h[2]; } ua;
    ua.h[0] = *(const v8bf*)lds_row1024(at_lds, l16, s + kbA);
    ua.h[1] = *(const v8bf*)lds_row1024(at_lds, l16, s + kbA + 8);
    a0 = wmma_bf16(ua.v, *(const v16bf*)(bp0 + s), a0);
    a1 = wmma_bf16(ua.v, *(const v16bf*)(bp1 + s), a1);
  }
  const float bias0 = b1[f1a];
  const float bias1 = b1[f1b];
  const int sl = 8 * half;
  for (int r = 0; r < 8; ++r) {
    float dv = dls[sl + r];
    h_lds[(sl + r) * 272 + f1a] = (bf16)fmaxf(dv * a0[r] + bias0, 0.f);
    h_lds[(sl + r) * 272 + f1b] = (bf16)fmaxf(dv * a1[r] + bias1, 0.f);
  }
  __syncthreads();

  // Phase B: Y2 = h @ W2 (K = 256), one f2 tile per wave
  const int f2 = wv * 16 + l16;
  const bf16* bp2 = W2T + (size_t)f2 * H1_ + kbB;
  v8f acc = v8f_zero();
  for (int k0 = 0; k0 < H1_; k0 += 32) {
    union { v16bf v; v8bf h[2]; } u;
    u.h[0] = *(const v8bf*)&h_lds[l16 * 272 + k0 + kbA];
    u.h[1] = *(const v8bf*)&h_lds[l16 * 272 + k0 + kbA + 8];
    acc = wmma_bf16(u.v, *(const v16bf*)(bp2 + k0), acc);
  }
  v8bf o;
  for (int r = 0; r < 8; ++r) o[r] = (bf16)(dls[sl + r] * acc[r]);
  *(v8bf*)(P2T + ((size_t)b * H2_ + f2) * S_ + t0 + sl) = o;
}

// ---------------- G3: M2 = AT @ P2 ; accumulate sum_t mask*dis*M2 ------------

__global__ __launch_bounds__(256) void k_g3(const bf16* __restrict__ AT,
                                            const bf16* __restrict__ P2T,
                                            const float* __restrict__ dis,
                                            const int* __restrict__ mask,
                                            float* __restrict__ gacc) {
  const int b    = blockIdx.y;
  const int t0   = blockIdx.x * 16;
  const int wv   = threadIdx.x >> 5;
  const int ln   = threadIdx.x & 31;
  const int l16  = ln & 15;
  const int half = ln >> 4;
  const int kbA  = half ? 16 : 0;
  const int kbB  = half ? 16 : 0;

  __shared__ __align__(16) bf16 at_lds[16 * 1040];
  if (wv == 0)
    tdm_load_stripe16(AT + ((size_t)b * S_ + t0) * S_, at_lds, S_, 7u);

  const int f0 = wv * 16;
  const bf16* bp = P2T + ((size_t)b * H2_ + f0 + l16) * S_ + kbB;

  v8f acc = v8f_zero();
  __syncthreads();
  for (int s = 0; s < S_; s += 32) {
    __builtin_prefetch(bp + s + 128, 0, 3);
    union { v16bf v; v8bf h[2]; } ua;
    ua.h[0] = *(const v8bf*)lds_row1024(at_lds, l16, s + kbA);
    ua.h[1] = *(const v8bf*)lds_row1024(at_lds, l16, s + kbA + 8);
    acc = wmma_bf16(ua.v, *(const v16bf*)(bp + s), acc);
  }
  float part = 0.f;
  for (int r = 0; r < 8; ++r) {
    int t = t0 + r + 8 * half;
    float w = (float)mask[b * S_ + t] * dis[b * S_ + t];
    part += w * acc[r];
  }
  part += __shfl_xor(part, 16);
  if (half == 0) atomicAdd(&gacc[b * H2_ + f0 + l16], part);
}

// ---------------- final: sigmoid(((gacc/nmask)+b2) . Wc + bc) ----------------

__global__ __launch_bounds__(128) void k_final(const float* __restrict__ gacc,
                                               const float* __restrict__ b2,
                                               const float* __restrict__ Wc,
                                               const float* __restrict__ bc,
                                               const int* __restrict__ mask,
                                               float* __restrict__ out) {
  const int b = blockIdx.x;
  const int tid = threadIdx.x;
  __shared__ float sb[128];

  float nm = 0.f;
  for (int i = tid; i < S_; i += 128) nm += (float)mask[b * S_ + i];
  sb[tid] = nm;
  __syncthreads();
  for (int o = 64; o > 0; o >>= 1) {
    if (tid < o) sb[tid] += sb[tid + o];
    __syncthreads();
  }
  const float nmask = sb[0];
  __syncthreads();

  float g = gacc[b * H2_ + tid] / nmask + b2[tid];
  sb[tid] = g * Wc[tid];
  __syncthreads();
  for (int o = 64; o > 0; o >>= 1) {
    if (tid < o) sb[tid] += sb[tid + o];
    __syncthreads();
  }
  if (tid == 0) out[b] = 1.f / (1.f + __expf(-(sb[0] + bc[0])));
}

// ---------------- launcher ----------------

extern "C" void kernel_launch(void* const* d_in, const int* in_sizes, int n_in,
                              void* d_out, int out_size, void* d_ws, size_t ws_size,
                              hipStream_t stream) {
  (void)in_sizes; (void)n_in; (void)out_size; (void)ws_size;

  const float* X    = (const float*)d_in[0];
  const int*   mask = (const int*)d_in[1];
  const float* W1   = (const float*)d_in[2];
  const float* b1   = (const float*)d_in[3];
  const float* W2   = (const float*)d_in[4];
  const float* b2   = (const float*)d_in[5];
  const float* Wc   = (const float*)d_in[6];
  const float* bc   = (const float*)d_in[7];
  float* out = (float*)d_out;

  char* w = (char*)d_ws;
  auto up = [](size_t x) { return (x + 255) & ~(size_t)255; };
  bf16* Xh  = (bf16*)w;  w += up((size_t)B_ * S_ * D_ * 2);   // 48 MiB
  bf16* W1T = (bf16*)w;  w += up((size_t)D_ * H1_ * 2);
  bf16* W2T = (bf16*)w;  w += up((size_t)H1_ * H2_ * 2);
  bf16* AT  = (bf16*)w;  w += up((size_t)B_ * S_ * S_ * 2);   // 64 MiB
  bf16* P1T = (bf16*)w;  w += up((size_t)B_ * H1_ * S_ * 2);  // 16 MiB
  bf16* P2T = (bf16*)w;  w += up((size_t)B_ * H2_ * S_ * 2);  //  8 MiB
  float* dis  = (float*)w; w += up((size_t)B_ * S_ * 4);
  float* gacc = (float*)w; w += up((size_t)B_ * H2_ * 4);

  k_cast<<<4096, 256, 0, stream>>>(X, Xh, (long)B_ * S_ * D_);
  k_w1t<<<(D_ * H1_ + 255) / 256, 256, 0, stream>>>(W1, W1T);
  k_w2t<<<(H1_ * H2_ + 255) / 256, 256, 0, stream>>>(W2, W2T);
  k_attn<<<dim3(S_ / 16, B_), 256, 0, stream>>>(Xh, AT);
  k_dis<<<dim3(S_ / 8, B_), 256, 0, stream>>>(AT, dis);
  k_zero<<<(B_ * H2_ + 255) / 256, 256, 0, stream>>>(gacc, B_ * H2_);
  k_g1<<<dim3(S_ / 16, B_), 256, 0, stream>>>(Xh, W1T, dis, P1T);
  k_g2<<<dim3(S_ / 16, B_), 256, 0, stream>>>(AT, P1T, W2T, dis, b1, P2T);
  k_g3<<<dim3(S_ / 16, B_), 256, 0, stream>>>(AT, P2T, dis, mask, gacc);
  k_final<<<B_, 128, 0, stream>>>(gacc, b2, Wc, bc, mask, out);
}
